// QRNNEncoderConcat_55370718380157
// MI455X (gfx1250) — compile-verified, matrix operand order checked by reading, not run on hardware
//
#include <hip/hip_runtime.h>

// ---------------------------------------------------------------------------
// QRNN encoder (3 layers) for gfx1250 (MI455X, wave32, WMMA).
// GEMM: bf16 inputs, f32 accumulate via v_wmma_f32_16x16x32_bf16, with
// double-buffered LDS tiles filled by GLOBAL_LOAD_ASYNC_TO_LDS (ASYNCcnt).
// Wave tile 64x48 (4x3 accumulators) for a high WMMA : ds_load ratio.
// Scan: f32 sequential recurrence per (b,h).
// ---------------------------------------------------------------------------

typedef __attribute__((ext_vector_type(16))) __bf16 v16bf;
typedef __attribute__((ext_vector_type(8)))  float  v8f;

#define QT 512
#define QB 64
#define QE 300
#define QH 800
#define QM (QT * QB)      // 32768 rows
#define QN (3 * QH)       // 2400 cols
#define TM 256
#define TN 96
#define TK 32

__device__ __forceinline__ unsigned lds_off32(const void* p) {
  // Generic (flat) pointers into LDS carry the DS byte offset in addr[31:0].
  return (unsigned)(unsigned long long)p;
}

__device__ __forceinline__ void async_ld_b128(unsigned lds_dst,
                                              const void* gsrc) {
  asm volatile("global_load_async_to_lds_b128 %0, %1, off"
               :
               : "v"(lds_dst), "v"(gsrc)
               : "memory");
}

__device__ __forceinline__ void async_ld_b64(unsigned lds_dst,
                                             const void* gsrc) {
  asm volatile("global_load_async_to_lds_b64 %0, %1, off"
               :
               : "v"(lds_dst), "v"(gsrc)
               : "memory");
}

// ---------------------------------------------------------------------------
// Weight transpose + pad + convert: W[f32, E x 3H] -> Wt[bf16, 3H x Kpad]
// ---------------------------------------------------------------------------
__global__ void qrnn_wt_kernel(const float* __restrict__ W,
                               __bf16* __restrict__ Wt,
                               int E, int Kpad) {
  int total = QN * Kpad;
  for (int idx = blockIdx.x * blockDim.x + threadIdx.x; idx < total;
       idx += gridDim.x * blockDim.x) {
    int n = idx / Kpad;
    int k = idx - n * Kpad;
    float v = (k < E) ? W[(size_t)k * QN + n] : 0.0f;
    Wt[idx] = (__bf16)v;
  }
}

// ---------------------------------------------------------------------------
// Input convert + pad: sent[f32, M x 300] -> X[bf16, M x 320]
// ---------------------------------------------------------------------------
__global__ void qrnn_cvt_in_kernel(const float* __restrict__ sent,
                                   __bf16* __restrict__ X) {
  const int Kpad = 320;
  long long total = (long long)QM * Kpad;
  for (long long idx = (long long)blockIdx.x * blockDim.x + threadIdx.x;
       idx < total; idx += (long long)gridDim.x * blockDim.x) {
    long long row = idx / Kpad;
    int k = (int)(idx - row * Kpad);
    float v = (k < QE) ? sent[row * QE + k] : 0.0f;
    X[idx] = (__bf16)v;
  }
}

// ---------------------------------------------------------------------------
// GEMM: G[M x 2400](bf16) = X[M x K](bf16) * Wt[2400 x K]^T (bf16) + bias(f32)
// Block: 256 threads (8 waves). Tile 256x96; waves 4(M) x 2(N), each wave
// computes 64x48 via 4x3 WMMA accumulators (12 WMMAs per 14 ds_load_b128).
// Double-buffered LDS, async global->LDS staging (7 async issues per wave per
// stage: 4 x b128 for A, 3 x b64 for B), pipelined with s_wait_asynccnt.
// LDS rows permuted in 8-element chunks {0,2,1,3} so each lane's fragment is
// one contiguous 32B read matching the ISA 16-bit A/B VGPR layout.
// ---------------------------------------------------------------------------
__global__ __launch_bounds__(256) void qrnn_gemm_wmma_kernel(
    const __bf16* __restrict__ X, const __bf16* __restrict__ Wt,
    const float* __restrict__ bias, __bf16* __restrict__ G, int K) {
  __shared__ __align__(32) __bf16 ldsA[2][TM * TK];  // 2 x 16 KB
  __shared__ __align__(32) __bf16 ldsB[2][TN * TK];  // 2 x  6 KB

  const int tid   = threadIdx.x;
  const int wid   = tid >> 5;
  const int lane  = tid & 31;
  const int half  = (lane >> 4) & 1;
  const int l16   = lane & 15;
  const int mwave = (wid & 3) * 64;   // 4 waves along M, 64 rows each
  const int nwave = (wid >> 2) * 48;  // 2 waves along N, 48 cols each
  const size_t Mbase = (size_t)blockIdx.x * TM;
  const int    Nbase = blockIdx.y * TN;

  // Per-thread staging coordinates (uniform issue count per wave).
  // A tile: 1024 x b128 chunks -> 4 per thread (rows tid>>2 + {0,64,128,192}).
  const int ar  = tid >> 2;
  const int ac  = tid & 3;                            // k-chunk (8 elems)
  const int apc = ((ac & 1) << 1) | ((ac >> 1) & 1);  // perm {0,2,1,3}
  // B tile: 768 x b64 chunks -> 3 per thread. idx = tid + i*256.

  v8f acc[4][3];
#pragma unroll
  for (int m = 0; m < 4; ++m)
#pragma unroll
    for (int n = 0; n < 3; ++n)
      acc[m][n] = (v8f)0.0f;

  auto stage = [&](int k0, int buf) {
    // A tile: 4 async b128 per thread
#pragma unroll
    for (int i = 0; i < 4; ++i) {
      int r = ar + i * 64;
      async_ld_b128(lds_off32(&ldsA[buf][r * TK + apc * 8]),
                    X + (Mbase + r) * (size_t)K + k0 + ac * 8);
    }
    // B tile: 3 async b64 per thread
#pragma unroll
    for (int i = 0; i < 3; ++i) {
      int idx = tid + i * 256;
      int r = idx >> 3;
      int c4 = idx & 7;
      int c = c4 >> 1, sub = c4 & 1;
      int pc = ((c & 1) << 1) | ((c >> 1) & 1);
      async_ld_b64(lds_off32(&ldsB[buf][r * TK + pc * 8 + sub * 4]),
                   Wt + (size_t)(Nbase + r) * K + k0 + c4 * 4);
    }
  };

  auto compute = [&](int buf) {
    v16bf aF[4], bF[3];
#pragma unroll
    for (int m = 0; m < 4; ++m)
      aF[m] =
          *(const v16bf*)(&ldsA[buf][(mwave + m * 16 + l16) * TK + half * 16]);
#pragma unroll
    for (int n = 0; n < 3; ++n)
      bF[n] =
          *(const v16bf*)(&ldsB[buf][(nwave + n * 16 + l16) * TK + half * 16]);
#pragma unroll
    for (int m = 0; m < 4; ++m)
#pragma unroll
      for (int n = 0; n < 3; ++n)
        acc[m][n] = __builtin_amdgcn_wmma_f32_16x16x32_bf16(
            false, aF[m], false, bF[n], (short)0, acc[m][n], false, false);
  };

  stage(0, 0);

  int buf = 0;
  int k0 = 0;
  // Main pipelined loop: stage next tile, wait current, compute current.
  for (; k0 + TK < K; k0 += TK, buf ^= 1) {
    stage(k0 + TK, buf ^ 1);
    // 7 async issues just went out for the next buffer; ASYNCcnt <= 7 means
    // everything for the *current* buffer has landed in LDS.
    asm volatile("s_wait_asynccnt 0x7" ::: "memory");
    __syncthreads();  // all waves' transfers for `buf` visible
    compute(buf);
    __syncthreads();  // everyone done reading `buf` before it is refilled
  }
  // Peeled last iteration (nothing left to stage).
  asm volatile("s_wait_asynccnt 0x0" ::: "memory");
  __syncthreads();
  compute(buf);

  // Epilogue: bias add (f32), convert -> bf16 gates.
  // D layout: VGPR r holds M = r + 8*half, N = lane%16.
#pragma unroll
  for (int n = 0; n < 3; ++n) {
    const int col = Nbase + nwave + n * 16 + l16;
    const float bv = bias[col];
#pragma unroll
    for (int m = 0; m < 4; ++m) {
#pragma unroll
      for (int r = 0; r < 8; ++r) {
        size_t row = Mbase + mwave + m * 16 + r + 8 * half;
        G[row * (size_t)QN + col] = (__bf16)(acc[m][n][r] + bv);
      }
    }
  }
}

// ---------------------------------------------------------------------------
// fo-pool scan: per (b,h) thread, c_t = f*tanh(z) + (1-f)*c_{t-1}.
// Writes next-layer bf16 activations [M x 800] and final c into d_out.
// ---------------------------------------------------------------------------
__global__ __launch_bounds__(256) void qrnn_scan_kernel(
    const __bf16* __restrict__ G, __bf16* __restrict__ Xn,
    float* __restrict__ outp, int layer) {
  int idx = blockIdx.x * blockDim.x + threadIdx.x;  // 0 .. B*H-1
  if (idx >= QB * QH) return;
  int b = idx / QH;
  int h = idx - b * QH;

  float c = 0.0f;
  for (int t = 0; t < QT; ++t) {
    size_t base = ((size_t)t * QB + b) * QN;
    float z = (float)G[base + h];
    float f = (float)G[base + QH + h];
    float o = (float)G[base + 2 * QH + h];
    float zt = tanhf(z);
    float fs = 1.0f / (1.0f + __expf(-f));
    float os = 1.0f / (1.0f + __expf(-o));
    c = fs * zt + (1.0f - fs) * c;
    Xn[((size_t)t * QB + b) * QH + h] = (__bf16)(os * c);
  }
  outp[(size_t)b * QN + layer * QH + h] = c;
}

// ---------------------------------------------------------------------------
extern "C" void kernel_launch(void* const* d_in, const int* in_sizes, int n_in,
                              void* d_out, int out_size, void* d_ws,
                              size_t ws_size, hipStream_t stream) {
  (void)in_sizes; (void)n_in; (void)out_size; (void)ws_size;
  const float* sent = (const float*)d_in[0];
  // d_in[1] = lengths: unused by the reference.
  const float* W0 = (const float*)d_in[2];
  const float* b0 = (const float*)d_in[3];
  const float* W1 = (const float*)d_in[4];
  const float* b1 = (const float*)d_in[5];
  const float* W2 = (const float*)d_in[6];
  const float* b2 = (const float*)d_in[7];
  float* out = (float*)d_out;

  char* ws = (char*)d_ws;
  size_t off = 0;
  auto carve = [&](size_t bytes) -> char* {
    char* p = ws + off;
    off = (off + bytes + 255) & ~(size_t)255;
    return p;
  };
  __bf16* Wt = (__bf16*)carve((size_t)QN * QH * 2);        // 3.84 MB (max Kpad)
  __bf16* G  = (__bf16*)carve((size_t)QM * QN * 2);        // 157 MB
  __bf16* Xa = (__bf16*)carve((size_t)QM * QH * 2);        // 52 MB
  __bf16* Xb = (__bf16*)carve((size_t)QM * QH * 2);        // 52 MB (layer-0 X0 aliases here)

  const dim3 gemmGrid(QM / TM, QN / TN);  // 128 x 25
  const int scanBlocks = (QB * QH + 255) / 256;

  // ---- Layer 0 (K = 300 padded to 320) ----
  qrnn_wt_kernel<<<(QN * 320 + 255) / 256, 256, 0, stream>>>(W0, Wt, QE, 320);
  qrnn_cvt_in_kernel<<<(QM * 320) / 256, 256, 0, stream>>>(sent, Xb);
  qrnn_gemm_wmma_kernel<<<gemmGrid, 256, 0, stream>>>(Xb, Wt, b0, G, 320);
  qrnn_scan_kernel<<<scanBlocks, 256, 0, stream>>>(G, Xa, out, 0);

  // ---- Layer 1 (K = 800) ----
  qrnn_wt_kernel<<<(QN * QH + 255) / 256, 256, 0, stream>>>(W1, Wt, QH, QH);
  qrnn_gemm_wmma_kernel<<<gemmGrid, 256, 0, stream>>>(Xa, Wt, b1, G, QH);
  qrnn_scan_kernel<<<scanBlocks, 256, 0, stream>>>(G, Xb, out, 1);

  // ---- Layer 2 (K = 800) ----
  qrnn_wt_kernel<<<(QN * QH + 255) / 256, 256, 0, stream>>>(W2, Wt, QH, QH);
  qrnn_gemm_wmma_kernel<<<gemmGrid, 256, 0, stream>>>(Xb, Wt, b2, G, QH);
  qrnn_scan_kernel<<<scanBlocks, 256, 0, stream>>>(G, Xa, out, 2);
}
